// OuterProductMean_30391188586655
// MI455X (gfx1250) — matrix operand order repaired and checked by reading, not compiled, not run on hardware
//
#include <hip/hip_runtime.h>
#include <hip/hip_bf16.h>

// ---------------------------------------------------------------------------
// OuterProductMean for MI455X (gfx1250), wave32 + WMMA bf16 + async-to-LDS.
//   C_IN=256, C_H=32, C_Z=128, B=1, S=128, N=256
// Pipeline:
//   k_prep : LayerNorm + left/right projections + mask -> bf16 packs
//   k_norm : rnorm[i][j] = 1 / max(sum_s m[s,i]*m[s,j], 1)
//   k_wot  : wo[cd][z] -> wo_t[z][cd] bf16
//   k_fused: per 16i x 16j block. A-fragments (whole i-strip) and wo
//            fragments live in registers for the entire kernel; per-j B
//            tiles double-buffered via global_load_async_to_lds_b128.
// ---------------------------------------------------------------------------

typedef __attribute__((ext_vector_type(16))) __bf16 v16bf;
typedef __attribute__((ext_vector_type(8)))  float  v8f;
typedef unsigned short ushort_t;

union Frag {               // 32 bytes: one WMMA 16x32 bf16 operand per lane
    uint4 u[2];
    v16bf v;
};

__device__ __forceinline__ ushort_t f2bf(float x) {
    unsigned int u = __float_as_uint(x);
    unsigned int r = u + 0x7FFFu + ((u >> 16) & 1u);   // round-to-nearest-even
    return (ushort_t)(r >> 16);
}

#define S_DIM   128
#define N_DIM   256
#define C_IN    256
#define C_H     32
#define C_Z     128
#define CD      (C_H * C_H)     // 1024
#define BROW    136             // 128 + 8 bf16 pad  -> conflict-free ds_load_b128
#define MROW    1032            // 1024 + 8 bf16 pad -> conflict-free ds_load_b128

// ---------------------------------------------------------------------------
// Kernel 1: one wave per (s,i) row. LN over 256, two 256->32 projections,
// mask, pack bf16 as [i|j][c|d][s].
// ---------------------------------------------------------------------------
__global__ __launch_bounds__(256)
void k_prep(const float* __restrict__ msa, const float* __restrict__ mask,
            const float* __restrict__ ln_w, const float* __restrict__ ln_b,
            const float* __restrict__ wl, const float* __restrict__ bl,
            const float* __restrict__ wr, const float* __restrict__ br,
            ushort_t* __restrict__ Apack, ushort_t* __restrict__ Bpack) {
    __shared__ float xrow[8][C_IN];
    const int wave = threadIdx.x >> 5;
    const int lane = threadIdx.x & 31;
    const int row  = blockIdx.x * 8 + wave;      // row = s*256 + i
    const int s    = row >> 8;
    const int i    = row & 255;

    const float* xp = msa + (size_t)row * C_IN;
    float v[8];
    float sum = 0.f, sq = 0.f;
    const int k0 = lane * 8;
#pragma unroll
    for (int t = 0; t < 8; ++t) { v[t] = xp[k0 + t]; sum += v[t]; sq += v[t] * v[t]; }
#pragma unroll
    for (int off = 16; off > 0; off >>= 1) {
        sum += __shfl_xor(sum, off, 32);
        sq  += __shfl_xor(sq,  off, 32);
    }
    const float mu   = sum * (1.0f / C_IN);
    const float var  = sq * (1.0f / C_IN) - mu * mu;
    const float rstd = rsqrtf(var + 1e-5f);
#pragma unroll
    for (int t = 0; t < 8; ++t)
        xrow[wave][k0 + t] = (v[t] - mu) * rstd * ln_w[k0 + t] + ln_b[k0 + t];
    __syncthreads();

    const int c = lane;                           // 32 output channels, 1/lane
    float accA = bl[c], accB = br[c];
    for (int k = 0; k < C_IN; ++k) {
        const float xk = xrow[wave][k];
        accA = fmaf(xk, wl[k * C_H + c], accA);   // coalesced: lane-consecutive
        accB = fmaf(xk, wr[k * C_H + c], accB);
    }
    const float m = mask[row];
    Apack[(size_t)(i * C_H + c) * S_DIM + s] = f2bf(accA * m);
    Bpack[(size_t)(i * C_H + c) * S_DIM + s] = f2bf(accB * m);
}

// ---------------------------------------------------------------------------
// Kernel 2: rnorm[i][j] = 1 / clip(sum_s m[s,i] m[s,j], 1, inf)
// ---------------------------------------------------------------------------
__global__ __launch_bounds__(256)
void k_norm(const float* __restrict__ mask, float* __restrict__ rnorm) {
    const int i = blockIdx.x, j = threadIdx.x;
    float acc = 0.f;
    for (int s = 0; s < S_DIM; ++s)
        acc = fmaf(mask[s * N_DIM + i], mask[s * N_DIM + j], acc);
    rnorm[i * N_DIM + j] = 1.0f / fmaxf(acc, 1.0f);
}

// ---------------------------------------------------------------------------
// Kernel 3: wo[cd][z] fp32 -> wo_t[z][cd] bf16
// ---------------------------------------------------------------------------
__global__ __launch_bounds__(256)
void k_wot(const float* __restrict__ wo, ushort_t* __restrict__ wo_t) {
    const int idx = blockIdx.x * 256 + threadIdx.x;   // 131072 total
    const int cd = idx >> 7, z = idx & 127;
    wo_t[z * CD + cd] = f2bf(wo[cd * C_Z + z]);
}

// ---------------------------------------------------------------------------
// Kernel 4: fused outer-product GEMM + mean + projection GEMM.
// Grid (16 j-blocks, 16 i-blocks), 256 threads = 8 waves.
// All loop-invariant WMMA operands are register-resident:
//   afrag[t][ks] : stage-A A-matrix (16 frags, 128 VGPRs) - whole kernel
//   wofrag[ks]   : stage-B B-matrix (32 frags, 256 VGPRs) - whole kernel
//   bfrag[nt][ks]: stage-A B-matrix ( 8 frags,  64 VGPRs) - per j
// ---------------------------------------------------------------------------
__global__ __launch_bounds__(256)
void k_fused(const ushort_t* __restrict__ Apack, const ushort_t* __restrict__ Bpack,
             const ushort_t* __restrict__ wo_t, const float* __restrict__ rnorm,
             const float* __restrict__ bo, float* __restrict__ out) {
    __shared__ ushort_t Bbuf[2][C_H][BROW];   // 2 x 32 x 136 bf16 ~ 17.4 KB
    __shared__ ushort_t meanbuf[16 * MROW];   // 16 x 1032 bf16 ~ 33 KB

    const int tid  = threadIdx.x;
    const int wave = tid >> 5;
    const int lane = tid & 31;
    const int lm   = lane & 15;
    const int lhi  = lane >> 4;
    const int j0   = blockIdx.x * 16;
    const int i0   = blockIdx.y * 16;

    // ---- persistent stage-A A-fragments: entire 16-i strip, K=s=128 ----
    Frag afrag[16];                           // [t*4 + ks]
#pragma unroll
    for (int t = 0; t < 4; ++t) {
        const int rowg = i0 * C_H + (wave + 8 * t) * 16 + lm;   // flat i*32+c
#pragma unroll
        for (int ks = 0; ks < 4; ++ks) {
            const ushort_t* ap = Apack + (size_t)rowg * S_DIM + ks * 32 + lhi * 16;
            afrag[t * 4 + ks].u[0] = *(const uint4*)(ap);
            afrag[t * 4 + ks].u[1] = *(const uint4*)(ap + 8);
        }
    }
    // ---- persistent stage-B wo fragments: this wave's 16-z strip ----
    Frag wofrag[32];                          // [ks]
#pragma unroll
    for (int ks = 0; ks < 32; ++ks) {
        const ushort_t* bp = wo_t + (size_t)(wave * 16 + lm) * CD + ks * 32 + lhi * 16;
        wofrag[ks].u[0] = *(const uint4*)(bp);
        wofrag[ks].u[1] = *(const uint4*)(bp + 8);
    }

    // Async B-tile DMA into LDS (ASYNCcnt-tracked), double buffered.
    auto issue_b_tile = [&](int j, int buf) {
        const uint32_t bbase = (uint32_t)(size_t)(&Bbuf[buf][0][0]);
#pragma unroll
        for (int c = tid; c < 512; c += 256) {
            const int r8 = c >> 4, c8 = c & 15;
            const uint32_t dst = bbase + (uint32_t)(r8 * BROW + c8 * 8) * 2u;
            const ushort_t* src = Bpack + (size_t)j * (C_H * S_DIM) + c * 8;
            asm volatile("global_load_async_to_lds_b128 %0, %1, off"
                         :: "v"(dst), "v"(src) : "memory");
        }
    };
    issue_b_tile(j0, 0);

    for (int jj = 0; jj < 16; ++jj) {
        const int j   = j0 + jj;
        const int buf = jj & 1;

        // retire this wave's async tile loads, then make visible block-wide
#if __has_builtin(__builtin_amdgcn_s_wait_asynccnt)
        __builtin_amdgcn_s_wait_asynccnt(0);
#else
        asm volatile("s_wait_asynccnt 0x0" ::: "memory");
#endif
        __syncthreads();

        // prefetch next tile into the other buffer (hidden behind ~64 WMMAs)
        if (jj < 15) issue_b_tile(j + 1, buf ^ 1);

        // ---- per-j B fragments from LDS (conflict-free b128) ----
        Frag bfrag[8];                        // [nt*4 + ks]
#pragma unroll
        for (int nt = 0; nt < 2; ++nt)
#pragma unroll
            for (int ks = 0; ks < 4; ++ks) {
                const ushort_t* bp = &Bbuf[buf][nt * 16 + lm][ks * 32 + lhi * 16];
                bfrag[nt * 4 + ks].u[0] = *(const uint4*)(bp);
                bfrag[nt * 4 + ks].u[1] = *(const uint4*)(bp + 8);
            }

        // ---- stage A: O[(ii,c), d] = sum_s A * B ; M=512,N=32,K=128 ----
        __bf16* mb = (__bf16*)meanbuf;
#pragma unroll
        for (int t = 0; t < 4; ++t) {
            const int mt = wave + 8 * t;              // M-tile 0..31
            const int ii = mt >> 1;
            const float rn = rnorm[(i0 + ii) * N_DIM + j];
#pragma unroll
            for (int nt = 0; nt < 2; ++nt) {
                v8f acc = {0.f, 0.f, 0.f, 0.f, 0.f, 0.f, 0.f, 0.f};
#pragma unroll
                for (int ks = 0; ks < 4; ++ks)
                    acc = __builtin_amdgcn_wmma_f32_16x16x32_bf16(
                              false, afrag[t * 4 + ks].v,
                              false, bfrag[nt * 4 + ks].v,
                              (short)0, acc, false, false);
                const int d = nt * 16 + lm;
#pragma unroll
                for (int r = 0; r < 8; ++r) {
                    const int c = (mt & 1) * 16 + r + lhi * 8;
                    mb[ii * MROW + c * C_H + d] = (__bf16)(acc[r] * rn);
                }
            }
        }
        __syncthreads();

        // ---- stage B: out[ii, z] = mean[ii, :] @ wo ; M=16,K=1024,N=128 ----
        {
            v8f acc = {0.f, 0.f, 0.f, 0.f, 0.f, 0.f, 0.f, 0.f};
#pragma unroll
            for (int ks = 0; ks < 32; ++ks) {
                Frag a;
                const ushort_t* ap = meanbuf + lm * MROW + ks * 32 + lhi * 16;
                a.u[0] = *(const uint4*)(ap);
                a.u[1] = *(const uint4*)(ap + 8);
                acc = __builtin_amdgcn_wmma_f32_16x16x32_bf16(
                          false, a.v, false, wofrag[ks].v, (short)0, acc,
                          false, false);
            }
            const int z = wave * 16 + lm;
            const float bz = bo[z];
#pragma unroll
            for (int r = 0; r < 8; ++r) {
                const int ii = r + lhi * 8;
                out[((size_t)(i0 + ii) * N_DIM + j) * C_Z + z] = acc[r] + bz;
            }
        }
        __syncthreads();
    }
}

// ---------------------------------------------------------------------------
extern "C" void kernel_launch(void* const* d_in, const int* in_sizes, int n_in,
                              void* d_out, int out_size, void* d_ws, size_t ws_size,
                              hipStream_t stream) {
    const float* msa  = (const float*)d_in[0];
    const float* mask = (const float*)d_in[1];
    const float* ln_w = (const float*)d_in[2];
    const float* ln_b = (const float*)d_in[3];
    const float* wl   = (const float*)d_in[4];
    const float* bl   = (const float*)d_in[5];
    const float* wr   = (const float*)d_in[6];
    const float* br   = (const float*)d_in[7];
    const float* wo   = (const float*)d_in[8];
    const float* bo   = (const float*)d_in[9];
    float* out = (float*)d_out;

    // workspace layout (bytes): Apack 2MB | Bpack 2MB | wo_t 256KB | rnorm 256KB
    char* ws = (char*)d_ws;
    ushort_t* Apack = (ushort_t*)(ws);
    ushort_t* Bpack = (ushort_t*)(ws + (2u << 20));
    ushort_t* wo_t  = (ushort_t*)(ws + (4u << 20));
    float*    rnp   = (float*)   (ws + (4u << 20) + (256u << 10));

    k_prep<<<dim3(4096), dim3(256), 0, stream>>>(msa, mask, ln_w, ln_b,
                                                 wl, bl, wr, br, Apack, Bpack);
    k_norm<<<dim3(256), dim3(256), 0, stream>>>(mask, rnp);
    k_wot<<<dim3(512), dim3(256), 0, stream>>>(wo, wo_t);
    k_fused<<<dim3(16, 16), dim3(256), 0, stream>>>(Apack, Bpack, wo_t, rnp,
                                                    bo, out);
}
